// SSIMLoss_19799799234910
// MI455X (gfx1250) — matrix operand order repaired and checked by reading, not compiled, not run on hardware
//
#include <hip/hip_runtime.h>

typedef _Float16 half8 __attribute__((ext_vector_type(8)));
typedef _Float16 v16h  __attribute__((ext_vector_type(16)));
typedef float    v8f   __attribute__((ext_vector_type(8)));

#define WIN        11
#define RAD        5
#define IMG        512
#define TILE_X     16
#define TILE_Y     128
#define RAW_ROWS   144          // TILE_Y + 2*RAD, rounded to 16-row blocks
#define RAW_STRIDE 40           // 32 data cols + 8 pad halves -> conflict-free b128
#define HT_STRIDE  152          // 144 rows + 8 pad halves -> conflict-free b128
#define C1_CONST   0.0001f     // 0.01^2
#define C2_CONST   0.0009f     // 0.03^2

union V16 { v16h v; _Float16 h[16]; };
union V8H { half8 v; _Float16 h[8]; };

// Build the A-matrix fragment for field f from centered p/t data.
// f: 0=p, 1=t, 2=p*p, 3=t*t, 4=p*t
__device__ __forceinline__ v16h buildA(int f, half8 plo, half8 pup,
                                       half8 tlo, half8 tup) {
    half8 alo, aup;
    if (f == 0)      { alo = plo;       aup = pup; }
    else if (f == 1) { alo = tlo;       aup = tup; }
    else if (f == 2) { alo = plo * plo; aup = pup * pup; }
    else if (f == 3) { alo = tlo * tlo; aup = tup * tup; }
    else             { alo = plo * tlo; aup = pup * tup; }
    return __builtin_shufflevector(alo, aup,
        0,1,2,3,4,5,6,7,8,9,10,11,12,13,14,15);
}

__global__ __launch_bounds__(256)
void ssim_main(const float* __restrict__ pred, const float* __restrict__ targ,
               float* __restrict__ partial) {
    __shared__ alignas(16) _Float16 sP[RAW_ROWS * RAW_STRIDE];
    __shared__ alignas(16) _Float16 sT[RAW_ROWS * RAW_STRIDE];
    __shared__ alignas(16) _Float16 sH[5 * 16 * HT_STRIDE];   // [field][col][row]
    __shared__ float sG[WIN];
    __shared__ float sRed[256];

    const int tid  = threadIdx.x;
    const int lane = tid & 31;
    const int n16  = lane & 15;         // column within tile / A-row M
    const int hi   = (lane >> 4) & 1;   // lane half select
    // Wave id as a proven-uniform scalar so work assignment is SALU branches.
    const int w    = __builtin_amdgcn_readfirstlane(tid >> 5);

    const int bx  = blockIdx.x;
    const int xt  = bx & 31;            // 32 x-tiles of 16
    const int yt  = (bx >> 5) & 3;      // 4 y-tiles of 128
    const int img = bx >> 7;            // 64 images
    const int X   = xt * TILE_X;
    const int Y   = yt * TILE_Y;
    const size_t imgOff = (size_t)img * IMG * IMG;

    // ---- Gaussian taps (sigma = 1.5), normalized ----
    if (tid < WIN) {
        float d = (float)(tid - RAD);
        sG[tid] = expf(-(d * d) / 4.5f);
    }
    __syncthreads();
    float gsum = 0.f;
    for (int k = 0; k < WIN; ++k) gsum += sG[k];
    const float ginv = 1.0f / gsum;

    // ---- Stage 0: raw tiles -> LDS as centered f16 (padding value = -0.5) ----
    for (int i = tid; i < RAW_ROWS * 32; i += 256) {
        int r  = i >> 5;
        int j  = i & 31;
        int gr = Y - RAD + r;
        int gc = X - RAD + j;
        float pv = -0.5f, tv = -0.5f;
        if (gr >= 0 && gr < IMG && gc >= 0 && gc < IMG) {
            size_t off = imgOff + (size_t)gr * IMG + (size_t)gc;
            pv = pred[off] - 0.5f;
            tv = targ[off] - 0.5f;
        }
        sP[r * RAW_STRIDE + j] = (_Float16)pv;
        sT[r * RAW_STRIDE + j] = (_Float16)tv;
    }

    // ---- Constant band-matrix fragments ----
    // Sh as B-matrix 32x16: lanes 0-15 hold K={0..7,16..23}, lanes 16-31 K={8..15,24..31}; N = lane%16.
    V16 shB;
    {
        int kb = hi ? 8 : 0;
        for (int e = 0; e < 16; ++e) {
            int k = (kb + e + (e >= 8 ? 8 : 0)) - n16;                 // tap index = j - n
            shB.h[e] = (_Float16)((k >= 0 && k < WIN) ? sG[k] * ginv : 0.f);
        }
    }
    // Sv as A-matrix 16x32: lane<16: M=lane, K={0..7,16..23}; lane>=16: M=lane-16, K={8..15,24..31}.
    V16 svA;
    {
        for (int e = 0; e < 8; ++e) {
            int K0 = (hi ? 8 : 0) + e;
            int K1 = (hi ? 24 : 16) + e;
            int k0 = K0 - n16, k1 = K1 - n16;
            svA.h[e]     = (_Float16)((k0 >= 0 && k0 < WIN) ? sG[k0] * ginv : 0.f);
            svA.h[e + 8] = (_Float16)((k1 >= 0 && k1 < WIN) ? sG[k1] * ginv : 0.f);
        }
    }
    __syncthreads();

    // ---- Stage 1: horizontal pass (9 row-blocks x 5 fields = 45 WMMAs) ----
    // H[hr][n] = sum_k g[k] * field[hr][n+k]; stored transposed in sH[f][n][hr].
    const int kbA = hi ? 8 : 0;   // A-matrix K base for this lane half
    {
        // Main row-block: one per wave.
        const int rb  = w;
        const int row = rb * 16 + n16;
        const _Float16* sp = sP + row * RAW_STRIDE + kbA;
        const _Float16* st = sT + row * RAW_STRIDE + kbA;
        half8 plo = ((const half8*)sp)[0], pup = ((const half8*)sp)[2];
        half8 tlo = ((const half8*)st)[0], tup = ((const half8*)st)[2];
#pragma unroll
        for (int f = 0; f < 5; ++f) {
            v16h A = buildA(f, plo, pup, tlo, tup);
            v8f c = {};
            c = __builtin_amdgcn_wmma_f32_16x16x32_f16(false, A, false, shB.v,
                                                       (short)0, c, false, false);
            // C layout: VGPR i -> row (i + 8*hi) of the block, col = n16.
            V8H out;
#pragma unroll
            for (int i = 0; i < 8; ++i) out.h[i] = (_Float16)c[i];
            _Float16* dst = sH + f * (16 * HT_STRIDE) + n16 * HT_STRIDE
                          + rb * 16 + 8 * hi;
            *(half8*)dst = out.v;
        }
    }
    if (w < 5) {
        // Tail row-block 8: its 5 fields spread over waves 0..4 (scalar branch).
        const int rb  = 8;
        const int f   = w;
        const int row = rb * 16 + n16;
        const _Float16* sp = sP + row * RAW_STRIDE + kbA;
        const _Float16* st = sT + row * RAW_STRIDE + kbA;
        half8 plo = ((const half8*)sp)[0], pup = ((const half8*)sp)[2];
        half8 tlo = ((const half8*)st)[0], tup = ((const half8*)st)[2];
        v16h A = buildA(f, plo, pup, tlo, tup);
        v8f c = {};
        c = __builtin_amdgcn_wmma_f32_16x16x32_f16(false, A, false, shB.v,
                                                   (short)0, c, false, false);
        V8H out;
#pragma unroll
        for (int i = 0; i < 8; ++i) out.h[i] = (_Float16)c[i];
        _Float16* dst = sH + f * (16 * HT_STRIDE) + n16 * HT_STRIDE
                      + rb * 16 + 8 * hi;
        *(half8*)dst = out.v;
    }
    __syncthreads();

    // ---- Stage 2: vertical pass + pointwise SSIM (1 output block per wave) ----
    {
        const int yb = w;    // 0..7
        v8f acc[5];
#pragma unroll
        for (int f = 0; f < 5; ++f) {
            // B-matrix 32x16 from transposed H: K = row offset (table order), N = col.
            const _Float16* src = sH + f * (16 * HT_STRIDE) + n16 * HT_STRIDE
                                + yb * 16 + (hi ? 8 : 0);
            half8 lo = ((const half8*)src)[0];
            half8 up = ((const half8*)src)[2];
            v16h B = __builtin_shufflevector(lo, up,
                0,1,2,3,4,5,6,7,8,9,10,11,12,13,14,15);
            v8f c = {};
            acc[f] = __builtin_amdgcn_wmma_f32_16x16x32_f16(false, svA.v, false, B,
                                                            (short)0, c, false, false);
        }
        // Centered means: mu_true = a + 0.5; sigmas are shift-invariant (exact incl. padding).
        float lsum = 0.f;
#pragma unroll
        for (int i = 0; i < 8; ++i) {
            float a   = acc[0][i];
            float b   = acc[1][i];
            float s1  = acc[2][i] - a * a;
            float s2  = acc[3][i] - b * b;
            float s12 = acc[4][i] - a * b;
            float A1 = a + 0.5f, B1 = b + 0.5f;
            float num = (2.f * A1 * B1 + C1_CONST) * (2.f * s12 + C2_CONST);
            float den = (A1 * A1 + B1 * B1 + C1_CONST) * (s1 + s2 + C2_CONST);
            lsum += num / den;
        }
        sRed[tid] = lsum;
    }
    __syncthreads();
    for (int s = 128; s > 0; s >>= 1) {
        if (tid < s) sRed[tid] += sRed[tid + s];
        __syncthreads();
    }
    if (tid == 0) partial[bx] = sRed[0];
}

__global__ __launch_bounds__(256)
void ssim_finalize(const float* __restrict__ partial, float* __restrict__ out, int nPart) {
    __shared__ float red[256];
    float s = 0.f;
    for (int i = threadIdx.x; i < nPart; i += 256) s += partial[i];
    red[threadIdx.x] = s;
    __syncthreads();
    for (int k = 128; k > 0; k >>= 1) {
        if (threadIdx.x < k) red[threadIdx.x] += red[threadIdx.x + k];
        __syncthreads();
    }
    if (threadIdx.x == 0)
        out[0] = 1.0f - red[0] * (1.0f / 16777216.0f);   // mean over 64*512*512
}

extern "C" void kernel_launch(void* const* d_in, const int* in_sizes, int n_in,
                              void* d_out, int out_size, void* d_ws, size_t ws_size,
                              hipStream_t stream) {
    const float* pred = (const float*)d_in[0];
    const float* targ = (const float*)d_in[1];
    float* partial = (float*)d_ws;          // 8192 floats of scratch
    const int nBlocks = 64 * 4 * 32;        // images * y-tiles * x-tiles = 8192
    ssim_main<<<nBlocks, 256, 0, stream>>>(pred, targ, partial);
    ssim_finalize<<<1, 256, 0, stream>>>(partial, (float*)d_out, nBlocks);
}
